// MultiHeadAttention_83133386982010
// MI455X (gfx1250) — matrix operand order repaired and checked by reading, not compiled, not run on hardware
//
#include <hip/hip_runtime.h>
#include <hip/hip_bf16.h>

// ---------------------------------------------------------------------------
// MI455X (gfx1250) multi-head causal attention, bf16 WMMA pipeline.
//   B=4, T=2048, D=512, H=8, head dim 64.
//   Stage 1: qkv[8192,1536](bf16) = x[8192,512](f32) @ w_qkv[1536,512]^T
//   Stage 2: flash attention per (b,h,64-row q tile), streaming softmax
//   Stage 3: out[8192,512](f32) = attn[8192,512](bf16) @ w_proj^T + b_proj
// Matrix math on v_wmma_f32_16x16x32_bf16; pure global->LDS copies use the
// CDNA5 async path (global_load_async_to_lds_b128 + ASYNCcnt).
// ---------------------------------------------------------------------------

typedef __attribute__((ext_vector_type(16))) __bf16 v16bf;
typedef __attribute__((ext_vector_type(8)))  __bf16 v8bf;
typedef __attribute__((ext_vector_type(4)))  __bf16 v4bf;
typedef __attribute__((ext_vector_type(8)))  float  v8f;

static __device__ __forceinline__ v16bf cat16(v8bf lo, v8bf hi) {
    return __builtin_shufflevector(lo, hi, 0,1,2,3,4,5,6,7,8,9,10,11,12,13,14,15);
}

static __device__ __forceinline__ v8f wmma_bf16(v16bf a, v16bf b, v8f c) {
    // (neg_a, A, neg_b, B, c_mod, C, reuse_a, reuse_b)
    return __builtin_amdgcn_wmma_f32_16x16x32_bf16(false, a, false, b, (short)0, c,
                                                   false, false);
}

// Async 16B global->LDS copy (per-lane). LDS flat aperture keeps the LDS byte
// address in addr[31:0] (ISA 10.2), so truncating the generic pointer gives
// the VDST operand. Tracked by ASYNCcnt.
static __device__ __forceinline__ void async_copy_b128(const void* gptr, void* lptr) {
    unsigned lds_addr = (unsigned)(uintptr_t)lptr;
    asm volatile("global_load_async_to_lds_b128 %0, %1, off"
                 :: "v"(lds_addr), "v"(gptr)
                 : "memory");
}

static __device__ __forceinline__ void wait_asynccnt0() {
#if __has_builtin(__builtin_amdgcn_s_wait_asynccnt)
    __builtin_amdgcn_s_wait_asynccnt(0);
#else
    asm volatile("s_wait_asynccnt 0x0" ::: "memory");
#endif
}

// ---------------------------------------------------------------------------
// Generic GEMM:  C[M,N] = A[M,K] * W[N,K]^T (+ bias)
// Block tile 128x128, K-step 32, 256 threads = 8 waves (4x2), wave tile 32x64.
// LDS tiles are bf16, padded to 40 elems/row (80B) -> conflict-free ds_load_b128
// fragment loads in both the ISA A-layout and B-layout.
// ---------------------------------------------------------------------------
#define BK    32
#define LDST  40   // 32 + 8 pad (bf16 elements)

template<bool A_BF16, bool OUT_BF16, bool BIAS>
__global__ __launch_bounds__(256)
void gemm_kernel(const void* __restrict__ Av, const float* __restrict__ W,
                 const float* __restrict__ bias, void* __restrict__ Cv,
                 int M, int N, int Kd)
{
    __shared__ alignas(16) __bf16 As[128 * LDST];
    __shared__ alignas(16) __bf16 Bs[128 * LDST];

    const int tid   = threadIdx.x;
    const int lane  = tid & 31;
    const int wave  = tid >> 5;
    const int waveM = wave >> 1;      // 0..3 -> 32-row strip
    const int waveN = wave & 1;       // 0..1 -> 64-col strip
    const int half  = lane >> 4;      // which 16-lane half
    const int l     = lane & 15;
    const int aOff  = half ? 8 : 0;   // ISA A-layout K offset for upper half
    const int m0    = blockIdx.x * 128;
    const int n0    = blockIdx.y * 128;

    v8f acc[2][4];
    #pragma unroll
    for (int mi = 0; mi < 2; mi++)
        #pragma unroll
        for (int ni = 0; ni < 4; ni++) acc[mi][ni] = (v8f){};

    for (int k0 = 0; k0 < Kd; k0 += BK) {
        // ---- stage A tile ----
        if constexpr (A_BF16) {
            // pure copy: CDNA5 async global->LDS, no VGPR round trip
            const __bf16* A = (const __bf16*)Av;
            #pragma unroll
            for (int i = 0; i < 2; i++) {
                int li = i * 256 + tid;
                int r = li >> 2, c = (li & 3) * 8;
                async_copy_b128(A + (size_t)(m0 + r) * Kd + k0 + c,
                                &As[r * LDST + c]);
            }
        } else {
            const float* A = (const float*)Av;
            #pragma unroll
            for (int i = 0; i < 4; i++) {
                int li = i * 256 + tid;
                int r = li >> 3, c = (li & 7) * 4;
                float4 v = *(const float4*)(A + (size_t)(m0 + r) * Kd + k0 + c);
                v4bf o = { (__bf16)v.x, (__bf16)v.y, (__bf16)v.z, (__bf16)v.w };
                *(v4bf*)(&As[r * LDST + c]) = o;
            }
        }
        // ---- stage W tile (always f32 weights, converted in VALU) ----
        #pragma unroll
        for (int i = 0; i < 4; i++) {
            int li = i * 256 + tid;
            int r = li >> 3, c = (li & 7) * 4;
            float4 v = *(const float4*)(W + (size_t)(n0 + r) * Kd + k0 + c);
            v4bf o = { (__bf16)v.x, (__bf16)v.y, (__bf16)v.z, (__bf16)v.w };
            *(v4bf*)(&Bs[r * LDST + c]) = o;
        }
        // prefetch next k-step of A and W toward L2
        if (k0 + BK < Kd) {
            if constexpr (A_BF16) {
                const __bf16* A = (const __bf16*)Av;
                __builtin_prefetch(A + (size_t)(m0 + (tid >> 1)) * Kd + k0 + BK, 0, 1);
            } else {
                const float* A = (const float*)Av;
                __builtin_prefetch(A + (size_t)(m0 + (tid >> 1)) * Kd + k0 + BK, 0, 1);
            }
            __builtin_prefetch(W + (size_t)(n0 + (tid >> 1)) * Kd + k0 + BK, 0, 1);
        }
        if constexpr (A_BF16) wait_asynccnt0();
        __syncthreads();

        // ---- fragments ----
        v16bf aF[2], bF[4];
        #pragma unroll
        for (int mi = 0; mi < 2; mi++) {
            const __bf16* p = &As[(waveM * 32 + mi * 16 + l) * LDST];
            aF[mi] = cat16(*(const v8bf*)(p + aOff),
                           *(const v8bf*)(p + aOff + 16));
        }
        #pragma unroll
        for (int ni = 0; ni < 4; ni++) {
            const __bf16* p = &Bs[(waveN * 64 + ni * 16 + l) * LDST + half * 8];
            bF[ni] = cat16(*(const v8bf*)p, *(const v8bf*)(p + 16));
        }
        #pragma unroll
        for (int mi = 0; mi < 2; mi++)
            #pragma unroll
            for (int ni = 0; ni < 4; ni++)
                acc[mi][ni] = wmma_bf16(aF[mi], bF[ni], acc[mi][ni]);
        __syncthreads();
    }

    // ---- epilogue: C layout -> lane holds col l(+half sel), VGPR j holds row ----
    #pragma unroll
    for (int mi = 0; mi < 2; mi++) {
        #pragma unroll
        for (int ni = 0; ni < 4; ni++) {
            int col = n0 + waveN * 64 + ni * 16 + l;
            float bv = 0.f;
            if constexpr (BIAS) bv = bias[col];
            #pragma unroll
            for (int j = 0; j < 8; j++) {
                int row = m0 + waveM * 32 + mi * 16 + half * 8 + j;
                float v = acc[mi][ni][j] + bv;
                if constexpr (OUT_BF16)
                    ((__bf16*)Cv)[(size_t)row * N + col] = (__bf16)v;
                else
                    ((float*)Cv)[(size_t)row * N + col] = v;
            }
        }
    }
}

// ---------------------------------------------------------------------------
// Flash attention. One block = (b, h, 64 q-rows); 4 waves x 16 q-rows each.
// Streams 64-key tiles up to the causal diagonal.
//   S = Q K^T  (wmma; K tile staged with async global->LDS copies)
//   online softmax with shfl_xor row reductions (rows live per half-wave)
//   O += P V   (V stored transposed in LDS; P round-trips through per-wave LDS
//               to convert C-layout f32 -> A-layout bf16)
// ---------------------------------------------------------------------------
#define LK 72   // 64 + 8 pad (bf16 elements)

__global__ __launch_bounds__(128)
void flash_kernel(const __bf16* __restrict__ qkv, __bf16* __restrict__ attnOut)
{
    constexpr int T = 2048, E = 1536, Dm = 512;
    __shared__ alignas(16) __bf16 Ks[64 * LK];       // [key][dim]
    __shared__ alignas(16) __bf16 Vt[64 * LK];       // [dim][key] (transposed)
    __shared__ alignas(16) __bf16 Ps[4][16 * LK];    // per-wave P staging

    const int tid  = threadIdx.x;
    const int lane = tid & 31;
    const int wave = tid >> 5;
    const int half = lane >> 4;
    const int l    = lane & 15;
    const int aOff = half ? 8 : 0;

    const int bid  = blockIdx.x;
    const int qt   = bid & 31;          // T/64 = 32 q tiles
    const int h    = (bid >> 5) & 7;
    const int b    = bid >> 8;
    const int qBase = qt * 64;
    const int qRowW = qBase + wave * 16;   // this wave's first q row

    // Q fragments held in registers for the whole key loop (A-layout).
    v16bf qF[2];
    {
        const __bf16* qp = qkv + (size_t)(b * T + qRowW + l) * E + h * 64;
        #pragma unroll
        for (int ks = 0; ks < 2; ks++)
            qF[ks] = cat16(*(const v8bf*)(qp + ks * 32 + aOff),
                           *(const v8bf*)(qp + ks * 32 + aOff + 16));
    }

    v8f o[4];
    #pragma unroll
    for (int ni = 0; ni < 4; ni++) o[ni] = (v8f){};
    float mprev[8], lsum[8];
    #pragma unroll
    for (int j = 0; j < 8; j++) { mprev[j] = -1e30f; lsum[j] = 0.f; }

    for (int kt = 0; kt <= qt; kt++) {
        const int kBase = kt * 64;

        // ---- stage K (async copy) and V (transposed, manual) tiles ----
        #pragma unroll
        for (int i = 0; i < 4; i++) {
            int c  = i * 128 + tid;
            int r  = c >> 3, cc = (c & 7) * 8;
            const __bf16* base = qkv + (size_t)(b * T + kBase + r) * E + h * 64;
            async_copy_b128(base + Dm + cc, &Ks[r * LK + cc]);
            v8bf vv = *(const v8bf*)(base + 2 * Dm + cc);
            #pragma unroll
            for (int j = 0; j < 8; j++) Vt[(cc + j) * LK + r] = vv[j];
        }
        // prefetch next key tile (K and V rows) toward L2
        if (kt < qt) {
            const __bf16* nbase =
                qkv + (size_t)(b * T + kBase + 64 + (tid >> 1)) * E + h * 64;
            __builtin_prefetch(nbase + Dm, 0, 1);
            __builtin_prefetch(nbase + 2 * Dm, 0, 1);
        }
        wait_asynccnt0();
        __syncthreads();

        // ---- S = Q K^T  (16 q-rows x 64 keys per wave) ----
        v8f s[4];
        #pragma unroll
        for (int ni = 0; ni < 4; ni++) s[ni] = (v8f){};
        #pragma unroll
        for (int ni = 0; ni < 4; ni++)
            #pragma unroll
            for (int ks = 0; ks < 2; ks++) {
                const __bf16* p = &Ks[(ni * 16 + l) * LK + ks * 32 + half * 8];
                v16bf bF = cat16(*(const v8bf*)p, *(const v8bf*)(p + 16));
                s[ni] = wmma_bf16(qF[ks], bF, s[ni]);
            }

        // ---- scale + causal mask + online softmax ----
        const bool diag = (kt == qt);
        #pragma unroll
        for (int j = 0; j < 8; j++) {
            const int qi = qRowW + half * 8 + j;
            float m = -1e30f;
            #pragma unroll
            for (int ni = 0; ni < 4; ni++) {
                float v = s[ni][j] * 0.125f;            // 1/sqrt(64)
                if (diag) {
                    int ki = kBase + ni * 16 + l;
                    if (ki > qi) v = -1e30f;
                }
                s[ni][j] = v;
                m = fmaxf(m, v);
            }
            // row max over the 16 lanes of this half-wave
            m = fmaxf(m, __shfl_xor(m, 1, 32));
            m = fmaxf(m, __shfl_xor(m, 2, 32));
            m = fmaxf(m, __shfl_xor(m, 4, 32));
            m = fmaxf(m, __shfl_xor(m, 8, 32));
            float mn    = fmaxf(mprev[j], m);
            float alpha = __expf(mprev[j] - mn);
            float rs = 0.f;
            #pragma unroll
            for (int ni = 0; ni < 4; ni++) {
                float p = __expf(s[ni][j] - mn);
                s[ni][j] = p;
                rs += p;
            }
            rs += __shfl_xor(rs, 1, 32);
            rs += __shfl_xor(rs, 2, 32);
            rs += __shfl_xor(rs, 4, 32);
            rs += __shfl_xor(rs, 8, 32);
            lsum[j]  = lsum[j] * alpha + rs;
            mprev[j] = mn;
            #pragma unroll
            for (int ni = 0; ni < 4; ni++) o[ni][j] *= alpha;
            // park P (bf16) in this wave's LDS patch at its C-layout position
            #pragma unroll
            for (int ni = 0; ni < 4; ni++)
                Ps[wave][(half * 8 + j) * LK + ni * 16 + l] = (__bf16)s[ni][j];
        }
        // same-wave LDS store->load; make the reload explicit (CDNA5 split cnt)
        asm volatile("s_wait_dscnt 0x0" ::: "memory");

        // ---- O += P V  (P reloaded in A-layout, V^T gives contiguous B-frag) ----
        #pragma unroll
        for (int ks = 0; ks < 2; ks++) {
            const __bf16* pp = &Ps[wave][l * LK + ks * 32];
            v16bf aP = cat16(*(const v8bf*)(pp + aOff),
                             *(const v8bf*)(pp + aOff + 16));
            #pragma unroll
            for (int ni = 0; ni < 4; ni++) {
                const __bf16* vp = &Vt[(ni * 16 + l) * LK + ks * 32 + half * 8];
                v16bf bV = cat16(*(const v8bf*)vp, *(const v8bf*)(vp + 16));
                o[ni] = wmma_bf16(aP, bV, o[ni]);
            }
        }
        __syncthreads();   // protect Ks/Vt for next tile
    }

    // ---- normalize and emit [B,T,D] bf16 for the output projection ----
    #pragma unroll
    for (int j = 0; j < 8; j++) {
        float inv = 1.0f / lsum[j];
        int row = qRowW + half * 8 + j;
        #pragma unroll
        for (int ni = 0; ni < 4; ni++) {
            int col = h * 64 + ni * 16 + l;
            attnOut[(size_t)(b * T + row) * Dm + col] = (__bf16)(o[ni][j] * inv);
        }
    }
}

// ---------------------------------------------------------------------------
extern "C" void kernel_launch(void* const* d_in, const int* in_sizes, int n_in,
                              void* d_out, int out_size, void* d_ws, size_t ws_size,
                              hipStream_t stream)
{
    (void)in_sizes; (void)n_in; (void)out_size; (void)ws_size;
    const float* x      = (const float*)d_in[0];   // [4,2048,512]
    const float* w_qkv  = (const float*)d_in[1];   // [1536,512]
    const float* w_proj = (const float*)d_in[2];   // [512,512]
    const float* b_proj = (const float*)d_in[3];   // [512]
    float* out = (float*)d_out;                    // [4,2048,512] f32

    constexpr int Bb = 4, T = 2048, Dm = 512, Hh = 8;
    constexpr int M = Bb * T;                      // 8192 rows

    // workspace: qkv bf16 [8192,1536] (25.2 MB) + attn bf16 [8192,512] (8.4 MB)
    __bf16* qkv  = (__bf16*)d_ws;
    __bf16* attn = qkv + (size_t)M * 3 * Dm;

    // 1) qkv = x @ w_qkv^T   (f32 in, bf16 out)
    gemm_kernel<false, true, false><<<dim3(M / 128, (3 * Dm) / 128), 256, 0, stream>>>(
        x, w_qkv, nullptr, qkv, M, 3 * Dm, Dm);

    // 2) causal flash attention over bf16 q/k/v
    flash_kernel<<<dim3(Bb * Hh * (T / 64)), 128, 0, stream>>>(qkv, attn);

    // 3) out = attn @ w_proj^T + b_proj   (bf16 in, f32 out)
    gemm_kernel<true, false, true><<<dim3(M / 128, Dm / 128), 256, 0, stream>>>(
        attn, w_proj, b_proj, out, M, Dm, Dm);
}
